// GptOssDecoderLayer_75041668596193
// MI455X (gfx1250) — compile-verified, matrix operand order checked by reading, not run on hardware
//
#include <hip/hip_runtime.h>
#include <hip/hip_bf16.h>
#include <cstdint>
#include <cstddef>

// ---------------------------------------------------------------------------
// GPT-OSS decoder layer for gfx1250 (MI455X), bf16 WMMA everywhere.
// 2x2 register-blocked GEMMs: 1 fragment load per v_wmma (16 flops/byte).
// ---------------------------------------------------------------------------

typedef __bf16 bf16;
typedef __attribute__((ext_vector_type(16))) __bf16 v16bf;
typedef __attribute__((ext_vector_type(8)))  float  v8f;

constexpr int   S_   = 2048;
constexpr int   H_   = 1024;
constexpr int   NH_  = 16;
constexpr int   NKV_ = 4;
constexpr int   HD_  = 64;
constexpr int   E_   = 8;
constexpr int   F_   = 2048;
constexpr float EPS_   = 1e-6f;
constexpr float ALPHA_ = 1.702f;
constexpr float LIMIT_ = 7.0f;

__device__ __forceinline__ v8f wmma_bf16(v16bf a, v16bf b, v8f c) {
    return __builtin_amdgcn_wmma_f32_16x16x32_bf16(
        /*neg_a=*/false, a, /*neg_b=*/false, b,
        /*c_mod=*/(short)0, c, /*reuse_a=*/false, /*reuse_b=*/false);
}

// A-fragment (16x32 bf16, M rows) and B^T-fragment (N rows x K) have identical
// per-lane layout: lane holds row (l&15); elements e<8 -> k = 8*half+e,
// e>=8 -> k = 8*half+8+e  (half = l>>4). Two contiguous 16B loads per lane.
__device__ __forceinline__ v16bf load_frag(const bf16* __restrict__ base,
                                           int row, int ld, int kbase) {
    int half = (threadIdx.x >> 4) & 1;
    const bf16* p = base + (size_t)row * ld + kbase + 8 * half;
    union { uint4 u[2]; v16bf v; } U;
    U.u[0] = *reinterpret_cast<const uint4*>(p);
    U.u[1] = *reinterpret_cast<const uint4*>(p + 16);
    return U.v;
}

// ---------------------------------------------------------------------------
// fp32 -> bf16 convert + transpose (dst[c*R + r] = src[r*C + c]), batched in z
// ---------------------------------------------------------------------------
__global__ void convtrans_kernel(const float* __restrict__ src, bf16* __restrict__ dst,
                                 int R, int C) {
    size_t total = (size_t)R * C;
    size_t base  = (size_t)blockIdx.z * total;
    for (size_t i = (size_t)blockIdx.x * blockDim.x + threadIdx.x; i < total;
         i += (size_t)gridDim.x * blockDim.x) {
        int r = (int)(i / C), c = (int)(i % C);
        dst[base + (size_t)c * R + r] = (bf16)src[base + i];
    }
}

// ---------------------------------------------------------------------------
// RMSNorm: one block per row, fp32 in, bf16 out
// ---------------------------------------------------------------------------
__global__ void rmsnorm_kernel(const float* __restrict__ x, const float* __restrict__ w,
                               bf16* __restrict__ out) {
    __shared__ float red[8];
    int t = blockIdx.x;
    const float* xr = x + (size_t)t * H_;
    float ss = 0.f;
    for (int i = threadIdx.x; i < H_; i += 256) { float v = xr[i]; ss += v * v; }
    #pragma unroll
    for (int m = 16; m >= 1; m >>= 1) ss += __shfl_xor(ss, m, 32);
    if ((threadIdx.x & 31) == 0) red[threadIdx.x >> 5] = ss;
    __syncthreads();
    float tot = 0.f;
    #pragma unroll
    for (int i = 0; i < 8; i++) tot += red[i];
    float scale = rsqrtf(tot / (float)H_ + EPS_);
    for (int i = threadIdx.x; i < H_; i += 256)
        out[(size_t)t * H_ + i] = (bf16)(xr[i] * scale * w[i]);
}

// ---------------------------------------------------------------------------
// Generic bf16 WMMA GEMM, 2x2 register blocking.
// block = 256 threads = 8 waves as (2 m) x (4 n); wave owns a 32x32 tile.
// C[M,N] = A[M,K] * BT[N,K]^T (+ optional residual).
// grid: x = N/128, y = M/64
// ---------------------------------------------------------------------------
__global__ void gemm_kernel(const bf16* __restrict__ A, const bf16* __restrict__ BT,
                            float* __restrict__ C, int N, int K,
                            int lda, int ldb, int ldc,
                            const float* __restrict__ residual) {
    int wave = threadIdx.x >> 5, lane = threadIdx.x & 31;
    int n = lane & 15, hi = lane >> 4;
    int mw = wave & 1, nw = wave >> 1;
    int mtile = blockIdx.y * 4 + mw * 2;              // 16-row tiles: mtile, mtile+1
    int ntile = (blockIdx.x * 4 + nw) * 2;            // 16-col tiles: ntile, ntile+1
    if (ntile * 16 >= N) return;                      // uniform per wave
    int arow0 = mtile * 16 + n, arow1 = arow0 + 16;
    int bcol0 = ntile * 16 + n, bcol1 = bcol0 + 16;
    v8f acc00 = {}, acc01 = {}, acc10 = {}, acc11 = {};
    for (int k = 0; k < K; k += 32) {
        // prefetch B-operand cachelines 8 k-steps ahead (global_prefetch_b8)
        __builtin_prefetch(BT + (size_t)bcol0 * ldb + k + 256, 0, 1);
        v16bf a0 = load_frag(A, arow0, lda, k);
        v16bf a1 = load_frag(A, arow1, lda, k);
        v16bf b0 = load_frag(BT, bcol0, ldb, k);
        v16bf b1 = load_frag(BT, bcol1, ldb, k);
        acc00 = wmma_bf16(a0, b0, acc00);
        acc01 = wmma_bf16(a0, b1, acc01);
        acc10 = wmma_bf16(a1, b0, acc10);
        acc11 = wmma_bf16(a1, b1, acc11);
    }
    #pragma unroll
    for (int mi = 0; mi < 2; mi++) {
        #pragma unroll
        for (int ni = 0; ni < 2; ni++) {
            v8f acc = (mi == 0) ? (ni == 0 ? acc00 : acc01)
                                : (ni == 0 ? acc10 : acc11);
            #pragma unroll
            for (int r = 0; r < 8; r++) {
                int rr = (mtile + mi) * 16 + r + 8 * hi;
                size_t idx = (size_t)rr * ldc + (ntile + ni) * 16 + n;
                float v = acc[r];
                if (residual) v += residual[idx];
                C[idx] = v;
            }
        }
    }
}

// ---------------------------------------------------------------------------
// RoPE + pack: qf/kf/vf fp32 -> q bf16 [S,NH*HD] (pre-scaled by 1/sqrt(HD)),
// k bf16 [NKV,S,HD], v^T bf16 [NKV,HD,S]
// ---------------------------------------------------------------------------
__global__ void rope_pack_kernel(const float* __restrict__ qf, const float* __restrict__ kf,
                                 const float* __restrict__ vf,
                                 const float* __restrict__ sins, const float* __restrict__ coss,
                                 bf16* __restrict__ qb, bf16* __restrict__ kb,
                                 bf16* __restrict__ vtb) {
    int s = blockIdx.x;
    for (int i = threadIdx.x; i < NH_ * HD_; i += blockDim.x) {
        int h = i >> 6, d = i & 63, dd = d & 31;
        float sv = sins[s * 32 + dd], cv = coss[s * 32 + dd];
        float xr = qf[(size_t)s * (NH_ * HD_) + h * 64 + dd];
        float xi = qf[(size_t)s * (NH_ * HD_) + h * 64 + dd + 32];
        float o  = (d < 32) ? (xr * cv - xi * sv) : (xr * sv + xi * cv);
        qb[(size_t)s * (NH_ * HD_) + i] = (bf16)(o * 0.125f);   // 1/sqrt(64)
    }
    for (int i = threadIdx.x; i < NKV_ * HD_; i += blockDim.x) {
        int h = i >> 6, d = i & 63, dd = d & 31;
        float sv = sins[s * 32 + dd], cv = coss[s * 32 + dd];
        float xr = kf[(size_t)s * (NKV_ * HD_) + h * 64 + dd];
        float xi = kf[(size_t)s * (NKV_ * HD_) + h * 64 + dd + 32];
        float o  = (d < 32) ? (xr * cv - xi * sv) : (xr * sv + xi * cv);
        kb [((size_t)h * S_ + s) * HD_ + d] = (bf16)o;
        vtb[((size_t)h * HD_ + d) * S_ + s] = (bf16)vf[(size_t)s * (NKV_ * HD_) + i];
    }
}

// ---------------------------------------------------------------------------
// Flash attention, transposed formulation (one wave per (16-query tile, head)):
//   S^T[key,query] = K * Q^T;  O^T[hd,query] = V^T * P^T
// D-fragment of S^T converts lane-for-lane into the B-fragment of P^T.
// ---------------------------------------------------------------------------
__global__ void attn_kernel(const bf16* __restrict__ qb, const bf16* __restrict__ kb,
                            const bf16* __restrict__ vtb, bf16* __restrict__ attnb) {
    int qtile = blockIdx.x, h = blockIdx.y;
    int lane = threadIdx.x & 31;
    int n = lane & 15, hi = lane >> 4;
    int qbase = qtile * 16;
    int kvh = h / (NH_ / NKV_);
    const bf16* Kh = kb  + (size_t)kvh * S_ * HD_;
    const bf16* Vh = vtb + (size_t)kvh * HD_ * S_;
    const bf16* Qh = qb  + (size_t)h * HD_;          // row stride NH*HD

    // Q B-fragments (reused across all key tiles), two hd chunks of 32
    v16bf bq0 = load_frag(Qh, qbase + n, NH_ * HD_, 0);
    v16bf bq1 = load_frag(Qh, qbase + n, NH_ * HD_, 32);

    float m_run = -1e30f, l_run = 0.f;
    v8f o0 = {}, o1 = {}, o2 = {}, o3 = {};
    int q_global = qbase + n;
    int qmax = qbase + 15;

    for (int kt = 0; kt <= qmax; kt += 32) {
        // A = K rows: two key subtiles x two hd chunks
        v16bf ak0a = load_frag(Kh, kt + n,      HD_, 0);
        v16bf ak0b = load_frag(Kh, kt + n,      HD_, 32);
        v16bf ak1a = load_frag(Kh, kt + 16 + n, HD_, 0);
        v16bf ak1b = load_frag(Kh, kt + 16 + n, HD_, 32);
        v8f s1 = {}, s2 = {};
        s1 = wmma_bf16(ak0a, bq0, s1);  s1 = wmma_bf16(ak0b, bq1, s1);
        s2 = wmma_bf16(ak1a, bq0, s2);  s2 = wmma_bf16(ak1b, bq1, s2);

        // causal mask + tile row-max (over keys, for this lane's query n)
        float tmax = -1e30f;
        #pragma unroll
        for (int r = 0; r < 8; r++) {
            int key1 = kt + r + 8 * hi;
            float v1 = (key1      <= q_global) ? s1[r] : -1e30f;
            float v2 = (key1 + 16 <= q_global) ? s2[r] : -1e30f;
            s1[r] = v1; s2[r] = v2;
            tmax = fmaxf(tmax, fmaxf(v1, v2));
        }
        tmax = fmaxf(tmax, __shfl_xor(tmax, 16, 32));
        float new_m = fmaxf(m_run, tmax);
        float corr  = __expf(m_run - new_m);

        // P^T in D-layout -> B-fragment: same lane, e<8 from s1, e>=8 from s2
        v16bf bp;
        float rsum = 0.f;
        #pragma unroll
        for (int r = 0; r < 8; r++) {
            float p1 = __expf(s1[r] - new_m); rsum += p1; bp[r]     = (bf16)p1;
            float p2 = __expf(s2[r] - new_m); rsum += p2; bp[8 + r] = (bf16)p2;
        }
        rsum += __shfl_xor(rsum, 16, 32);
        l_run = l_run * corr + rsum;
        m_run = new_m;
        #pragma unroll
        for (int r = 0; r < 8; r++) { o0[r] *= corr; o1[r] *= corr; o2[r] *= corr; o3[r] *= corr; }

        // O^T += V^T * P^T : A = V^T rows (hd), K = 32 keys
        v16bf av;
        av = load_frag(Vh,  0 + n, S_, kt); o0 = wmma_bf16(av, bp, o0);
        av = load_frag(Vh, 16 + n, S_, kt); o1 = wmma_bf16(av, bp, o1);
        av = load_frag(Vh, 32 + n, S_, kt); o2 = wmma_bf16(av, bp, o2);
        av = load_frag(Vh, 48 + n, S_, kt); o3 = wmma_bf16(av, bp, o3);
    }

    float inv = 1.f / l_run;                     // per-lane (its query n)
    size_t row = (size_t)(qbase + n) * (NH_ * HD_);
    #pragma unroll
    for (int r = 0; r < 8; r++) {
        int colb = h * 64 + r + 8 * hi;
        attnb[row + colb +  0] = (bf16)(o0[r] * inv);
        attnb[row + colb + 16] = (bf16)(o1[r] * inv);
        attnb[row + colb + 32] = (bf16)(o2[r] * inv);
        attnb[row + colb + 48] = (bf16)(o3[r] * inv);
    }
}

// ---------------------------------------------------------------------------
// Router: one wave per token. Top-2 softmax + atomic compaction into
// per-expert assignment lists (assignment id = 2*token + rank).
// ---------------------------------------------------------------------------
__global__ void zero_cnt_kernel(int* cnt) { if (threadIdx.x < E_) cnt[threadIdx.x] = 0; }

__global__ void router_kernel(const bf16* __restrict__ x2b, const float* __restrict__ wr,
                              const float* __restrict__ br, float* __restrict__ topw,
                              int* __restrict__ cnt, int* __restrict__ elist) {
    int t = blockIdx.x * (blockDim.x >> 5) + (threadIdx.x >> 5);
    int lane = threadIdx.x & 31;
    if (t >= S_) return;
    float acc[E_];
    #pragma unroll
    for (int e = 0; e < E_; e++) acc[e] = 0.f;
    for (int i = lane; i < H_; i += 32) {
        float x = (float)x2b[(size_t)t * H_ + i];
        #pragma unroll
        for (int e = 0; e < E_; e++) acc[e] += x * wr[i * E_ + e];
    }
    #pragma unroll
    for (int e = 0; e < E_; e++)
        #pragma unroll
        for (int m = 16; m >= 1; m >>= 1) acc[e] += __shfl_xor(acc[e], m, 32);
    if (lane == 0) {
        #pragma unroll
        for (int e = 0; e < E_; e++) acc[e] += br[e];
        int i0 = 0;
        #pragma unroll
        for (int e = 1; e < E_; e++) if (acc[e] > acc[i0]) i0 = e;
        int i1 = (i0 == 0) ? 1 : 0;
        #pragma unroll
        for (int e = 0; e < E_; e++) if (e != i0 && acc[e] > acc[i1]) i1 = e;
        float e1 = __expf(acc[i1] - acc[i0]);
        topw[2 * t]     = 1.f / (1.f + e1);
        topw[2 * t + 1] = e1  / (1.f + e1);
        int p0 = atomicAdd(&cnt[i0], 1); elist[i0 * S_ + p0] = 2 * t;
        int p1 = atomicAdd(&cnt[i1], 1); elist[i1 * S_ + p1] = 2 * t + 1;
    }
}

// ---------------------------------------------------------------------------
// MoE gate_up GEMM, 2x2 blocked, fused clipped-SwiGLU epilogue.
// block = 256 = 8 waves as (2 m) x (4 n); wave: 32 rows x 32 gate_up cols.
// grid: x = 2F/(8*32) = 16, y = S/64 = 32, z = expert
// ---------------------------------------------------------------------------
__global__ void moe_gateup_kernel(const bf16* __restrict__ x2b, const bf16* __restrict__ gupT,
                                  const float* __restrict__ gub, const int* __restrict__ elist,
                                  const int* __restrict__ cnt, bf16* __restrict__ gated) {
    int e = blockIdx.z;
    int count = cnt[e];
    int wave = threadIdx.x >> 5, lane = threadIdx.x & 31;
    int n = lane & 15, hi = lane >> 4;
    int mw = wave & 1, nw = wave >> 1;
    int rt0 = blockIdx.y * 4 + mw * 2;               // 16-row tiles rt0, rt0+1
    if (rt0 * 16 >= count) return;                   // uniform per wave
    int ridx0 = rt0 * 16 + n, ridx1 = ridx0 + 16;
    int a0 = elist[e * S_ + ((ridx0 < count) ? ridx0 : (count - 1))];
    int a1 = elist[e * S_ + ((ridx1 < count) ? ridx1 : (count - 1))];
    int tok0 = a0 >> 1, tok1 = a1 >> 1;
    int colbase = (blockIdx.x * 8 + nw) * 32;
    const bf16* BT = gupT + (size_t)e * (2 * F_) * H_;
    v8f acc00 = {}, acc01 = {}, acc10 = {}, acc11 = {};
    for (int k = 0; k < H_; k += 32) {
        __builtin_prefetch(BT + (size_t)(colbase + n) * H_ + k + 256, 0, 1);
        v16bf af0 = load_frag(x2b, tok0, H_, k);
        v16bf af1 = load_frag(x2b, tok1, H_, k);
        v16bf b0  = load_frag(BT, colbase + n,      H_, k);
        v16bf b1  = load_frag(BT, colbase + 16 + n, H_, k);
        acc00 = wmma_bf16(af0, b0, acc00);
        acc01 = wmma_bf16(af0, b1, acc01);
        acc10 = wmma_bf16(af1, b0, acc10);
        acc11 = wmma_bf16(af1, b1, acc11);
    }
    const float* bias = gub + (size_t)e * (2 * F_);
    #pragma unroll
    for (int mi = 0; mi < 2; mi++) {
        #pragma unroll
        for (int sub = 0; sub < 2; sub++) {
            v8f acc = (mi == 0) ? (sub == 0 ? acc00 : acc01)
                                : (sub == 0 ? acc10 : acc11);
            int col = colbase + sub * 16 + n;
            float b = bias[col];
            #pragma unroll
            for (int r = 0; r < 8; r++) {
                float v = acc[r] + b;
                float o = __shfl_xor(v, 1, 32);        // pair even/odd columns
                if ((col & 1) == 0) {
                    float g   = fminf(v, LIMIT_);
                    float u   = fminf(fmaxf(o, -LIMIT_), LIMIT_);
                    float glu = g / (1.f + __expf(-g * ALPHA_));
                    float gg  = (u + 1.f) * glu;
                    int rowm = (rt0 + mi) * 16 + r + 8 * hi;
                    if (rowm < count) {
                        int aa = elist[e * S_ + rowm];
                        gated[(size_t)aa * F_ + (col >> 1)] = (bf16)gg;
                    }
                }
            }
        }
    }
}

// ---------------------------------------------------------------------------
// MoE down GEMM, 2x2 blocked: down_out[a,H] = gated[a,F]*downT[e]^T + bias
// grid: x = H/(8*32) = 4, y = S/64 = 32, z = expert; block 256
// ---------------------------------------------------------------------------
__global__ void moe_down_kernel(const bf16* __restrict__ gated, const bf16* __restrict__ downT,
                                const float* __restrict__ dwb, const int* __restrict__ elist,
                                const int* __restrict__ cnt, float* __restrict__ dout) {
    int e = blockIdx.z;
    int count = cnt[e];
    int wave = threadIdx.x >> 5, lane = threadIdx.x & 31;
    int n = lane & 15, hi = lane >> 4;
    int mw = wave & 1, nw = wave >> 1;
    int rt0 = blockIdx.y * 4 + mw * 2;
    if (rt0 * 16 >= count) return;
    int ridx0 = rt0 * 16 + n, ridx1 = ridx0 + 16;
    int a0 = elist[e * S_ + ((ridx0 < count) ? ridx0 : (count - 1))];
    int a1 = elist[e * S_ + ((ridx1 < count) ? ridx1 : (count - 1))];
    int colbase = (blockIdx.x * 8 + nw) * 32;
    const bf16* BT = downT + (size_t)e * H_ * F_;
    v8f acc00 = {}, acc01 = {}, acc10 = {}, acc11 = {};
    for (int k = 0; k < F_; k += 32) {
        __builtin_prefetch(BT + (size_t)(colbase + n) * F_ + k + 256, 0, 1);
        v16bf af0 = load_frag(gated, a0, F_, k);
        v16bf af1 = load_frag(gated, a1, F_, k);
        v16bf b0  = load_frag(BT, colbase + n,      F_, k);
        v16bf b1  = load_frag(BT, colbase + 16 + n, F_, k);
        acc00 = wmma_bf16(af0, b0, acc00);
        acc01 = wmma_bf16(af0, b1, acc01);
        acc10 = wmma_bf16(af1, b0, acc10);
        acc11 = wmma_bf16(af1, b1, acc11);
    }
    #pragma unroll
    for (int mi = 0; mi < 2; mi++) {
        #pragma unroll
        for (int ni = 0; ni < 2; ni++) {
            v8f acc = (mi == 0) ? (ni == 0 ? acc00 : acc01)
                                : (ni == 0 ? acc10 : acc11);
            #pragma unroll
            for (int r = 0; r < 8; r++) {
                int rowm = (rt0 + mi) * 16 + r + 8 * hi;
                if (rowm < count) {
                    int aa  = elist[e * S_ + rowm];
                    int col = colbase + ni * 16 + n;
                    dout[(size_t)aa * H_ + col] = acc[r] + dwb[(size_t)e * H_ + col];
                }
            }
        }
    }
}

// ---------------------------------------------------------------------------
// Final combine: out = hidden2 + w0*down_out[2t] + w1*down_out[2t+1]
// ---------------------------------------------------------------------------
__global__ void combine_kernel(const float* __restrict__ hidden2, const float* __restrict__ dout,
                               const float* __restrict__ topw, float* __restrict__ out) {
    size_t i = (size_t)blockIdx.x * blockDim.x + threadIdx.x;
    if (i >= (size_t)S_ * H_) return;
    size_t t = i / H_, c = i % H_;
    out[i] = hidden2[i]
           + topw[2 * t]     * dout[(2 * t)     * (size_t)H_ + c]
           + topw[2 * t + 1] * dout[(2 * t + 1) * (size_t)H_ + c];
}

// ---------------------------------------------------------------------------
extern "C" void kernel_launch(void* const* d_in, const int* in_sizes, int n_in,
                              void* d_out, int out_size, void* d_ws, size_t ws_size,
                              hipStream_t stream) {
    const float* hidden = (const float*)d_in[0];
    const float* sins   = (const float*)d_in[1];
    const float* coss   = (const float*)d_in[2];
    // d_in[3] = mask: causal handled analytically
    const float* ln1    = (const float*)d_in[4];
    const float* ln2    = (const float*)d_in[5];
    const float* wq     = (const float*)d_in[6];
    const float* wk     = (const float*)d_in[7];
    const float* wv     = (const float*)d_in[8];
    const float* wo     = (const float*)d_in[9];
    const float* wr     = (const float*)d_in[10];
    const float* br     = (const float*)d_in[11];
    const float* gup    = (const float*)d_in[12];
    const float* gub    = (const float*)d_in[13];
    const float* dwn    = (const float*)d_in[14];
    const float* dwb    = (const float*)d_in[15];
    float* out = (float*)d_out;

    char* ws = (char*)d_ws;
    size_t off = 0;
    auto take = [&](size_t bytes) -> char* {
        char* p = ws + off;
        off = (off + bytes + 255) & ~(size_t)255;
        return p;
    };
    bf16* wqT   = (bf16*)take((size_t)(NH_ * HD_) * H_ * 2);          // [1024,1024]
    bf16* wkT   = (bf16*)take((size_t)(NKV_ * HD_) * H_ * 2);         // [256,1024]
    bf16* wvT   = (bf16*)take((size_t)(NKV_ * HD_) * H_ * 2);
    bf16* woT   = (bf16*)take((size_t)H_ * (NH_ * HD_) * 2);          // [1024,1024]
    bf16* gupT  = (bf16*)take((size_t)E_ * (2 * F_) * H_ * 2);        // 64 MB
    bf16* dwnT  = (bf16*)take((size_t)E_ * H_ * F_ * 2);              // 32 MB
    bf16* xnorm = (bf16*)take((size_t)S_ * H_ * 2);
    float* qf   = (float*)take((size_t)S_ * NH_ * HD_ * 4);
    float* kf   = (float*)take((size_t)S_ * NKV_ * HD_ * 4);
    float* vf   = (float*)take((size_t)S_ * NKV_ * HD_ * 4);
    bf16* qb    = (bf16*)take((size_t)S_ * NH_ * HD_ * 2);
    bf16* kb    = (bf16*)take((size_t)NKV_ * S_ * HD_ * 2);
    bf16* vtb   = (bf16*)take((size_t)NKV_ * HD_ * S_ * 2);
    bf16* attnb = (bf16*)take((size_t)S_ * NH_ * HD_ * 2);
    float* hid2 = (float*)take((size_t)S_ * H_ * 4);
    bf16* x2b   = (bf16*)take((size_t)S_ * H_ * 2);
    float* topw = (float*)take((size_t)2 * S_ * 4);
    int*  cnt   = (int*)take(E_ * 4);
    int*  elist = (int*)take((size_t)E_ * S_ * 4);
    bf16* gated = (bf16*)take((size_t)2 * S_ * F_ * 2);               // 16 MB
    float* dmoe = (float*)take((size_t)2 * S_ * H_ * 4);              // 16 MB

    auto ctgrid = [](size_t total, int z) {
        unsigned gx = (unsigned)((total + 255) / 256);
        if (gx > 8192) gx = 8192;
        return dim3(gx, 1, (unsigned)z);
    };

    // ---- weight convert + transpose (fp32 -> bf16, [R,C] -> [C,R]) ----
    convtrans_kernel<<<ctgrid((size_t)H_ * 1024, 1), 256, 0, stream>>>(wq, wqT, H_, NH_ * HD_);
    convtrans_kernel<<<ctgrid((size_t)H_ * 256, 1),  256, 0, stream>>>(wk, wkT, H_, NKV_ * HD_);
    convtrans_kernel<<<ctgrid((size_t)H_ * 256, 1),  256, 0, stream>>>(wv, wvT, H_, NKV_ * HD_);
    convtrans_kernel<<<ctgrid((size_t)1024 * H_, 1), 256, 0, stream>>>(wo, woT, NH_ * HD_, H_);
    convtrans_kernel<<<ctgrid((size_t)H_ * 2 * F_, E_), 256, 0, stream>>>(gup, gupT, H_, 2 * F_);
    convtrans_kernel<<<ctgrid((size_t)F_ * H_, E_),     256, 0, stream>>>(dwn, dwnT, F_, H_);

    // ---- attention block ----
    rmsnorm_kernel<<<S_, 256, 0, stream>>>(hidden, ln1, xnorm);
    gemm_kernel<<<dim3((NH_ * HD_) / 128, S_ / 64), 256, 0, stream>>>(
        xnorm, wqT, qf, NH_ * HD_, H_, H_, H_, NH_ * HD_, nullptr);
    gemm_kernel<<<dim3((NKV_ * HD_) / 128, S_ / 64), 256, 0, stream>>>(
        xnorm, wkT, kf, NKV_ * HD_, H_, H_, H_, NKV_ * HD_, nullptr);
    gemm_kernel<<<dim3((NKV_ * HD_) / 128, S_ / 64), 256, 0, stream>>>(
        xnorm, wvT, vf, NKV_ * HD_, H_, H_, H_, NKV_ * HD_, nullptr);
    rope_pack_kernel<<<S_, 256, 0, stream>>>(qf, kf, vf, sins, coss, qb, kb, vtb);
    attn_kernel<<<dim3(S_ / 16, NH_), 32, 0, stream>>>(qb, kb, vtb, attnb);
    gemm_kernel<<<dim3(H_ / 128, S_ / 64), 256, 0, stream>>>(
        attnb, woT, hid2, H_, NH_ * HD_, NH_ * HD_, NH_ * HD_, H_, hidden);

    // ---- MoE block ----
    rmsnorm_kernel<<<S_, 256, 0, stream>>>(hid2, ln2, x2b);
    zero_cnt_kernel<<<1, 32, 0, stream>>>(cnt);
    router_kernel<<<S_ / 8, 256, 0, stream>>>(x2b, wr, br, topw, cnt, elist);
    moe_gateup_kernel<<<dim3((2 * F_) / 256, S_ / 64, E_), 256, 0, stream>>>(
        x2b, gupT, gub, elist, cnt, gated);
    moe_down_kernel<<<dim3(H_ / 256, S_ / 64, E_), 256, 0, stream>>>(
        gated, dwnT, dwb, elist, cnt, dmoe);
    combine_kernel<<<(unsigned)(((size_t)S_ * H_ + 255) / 256), 256, 0, stream>>>(
        hid2, dmoe, topw, out);
}
